// PointNetSetAbstraction_42880953483445
// MI455X (gfx1250) — matrix operand order repaired and checked by reading, not compiled, not run on hardware
//
#include <hip/hip_runtime.h>
#include <hip/hip_bf16.h>

typedef __attribute__((ext_vector_type(16))) _Float16 v16h;
typedef __attribute__((ext_vector_type(8)))  _Float16 v8h;
typedef __attribute__((ext_vector_type(8)))  float    v8f;

#define PN_B   16
#define PN_N   4096
#define PN_C   64
#define PN_S   1024     // NPOINT
#define PN_K   32       // NSAMPLE
#define PN_CIN 67       // 3 + C
#define PN_KP  96       // CIN padded to 3 * 32 for WMMA K-chunks
#define PN_M   128      // output channels
#define PN_R2  0.04f    // RADIUS^2

// ---------------------------------------------------------------------------
// Kernel 1: farthest point sampling. One block per batch, 1024 threads.
// dist[] lives in registers (4 points per thread); argmax via shfl + LDS tree.
// Emits fps_idx (int) and new_xyz (float3) exactly like the reference scan:
// iteration i records the *current* far, then computes the next one.
// ---------------------------------------------------------------------------
__global__ __launch_bounds__(1024)
void pn_fps_kernel(const float* __restrict__ xyz,
                   float* __restrict__ out_xyz,
                   int* __restrict__ out_idx)
{
    __shared__ float swv[32];
    __shared__ int   swi[32];
    __shared__ int   sfar;

    const int t = threadIdx.x;
    const int b = blockIdx.x;
    const int lane = t & 31;
    const int w = t >> 5;
    const float* X = xyz + (size_t)b * PN_N * 3;

    float dist[4] = {1e10f, 1e10f, 1e10f, 1e10f};
    if (t == 0) sfar = 0;
    __syncthreads();

    for (int i = 0; i < PN_S; ++i) {
        const int far = sfar;
        if (t == 0) {
            out_idx[b * PN_S + i] = far;
            out_xyz[(b * PN_S + i) * 3 + 0] = X[far * 3 + 0];
            out_xyz[(b * PN_S + i) * 3 + 1] = X[far * 3 + 1];
            out_xyz[(b * PN_S + i) * 3 + 2] = X[far * 3 + 2];
        }
        const float cx = X[far * 3 + 0];
        const float cy = X[far * 3 + 1];
        const float cz = X[far * 3 + 2];

        float bestv = -1.0f;
        int   besti = 0;
#pragma unroll
        for (int jj = 0; jj < 4; ++jj) {
            const int p = jj * 1024 + t;
            const float dx = X[p * 3 + 0] - cx;
            const float dy = X[p * 3 + 1] - cy;
            const float dz = X[p * 3 + 2] - cz;
            const float d = dx * dx + dy * dy + dz * dz;
            const float nd = fminf(dist[jj], d);
            dist[jj] = nd;
            if (nd > bestv) { bestv = nd; besti = p; }
        }
        // wave32 reduction (prefer lower index on ties, matching argmax)
#pragma unroll
        for (int off = 16; off >= 1; off >>= 1) {
            const float ov = __shfl_xor(bestv, off);
            const int   oi = __shfl_xor(besti, off);
            if (ov > bestv || (ov == bestv && oi < besti)) { bestv = ov; besti = oi; }
        }
        if (lane == 0) { swv[w] = bestv; swi[w] = besti; }
        __syncthreads();
        if (w == 0) {
            float v2 = swv[lane];
            int   i2 = swi[lane];
#pragma unroll
            for (int off = 16; off >= 1; off >>= 1) {
                const float ov = __shfl_xor(v2, off);
                const int   oi = __shfl_xor(i2, off);
                if (ov > v2 || (ov == v2 && oi < i2)) { v2 = ov; i2 = oi; }
            }
            if (lane == 0) sfar = i2;
        }
        __syncthreads();
    }
}

// ---------------------------------------------------------------------------
// Kernel 2: fold W3 @ W2 @ W1 -> W_eff (128 x 96, f16, zero-padded K) and
// b_eff = W3@W2@b1 + W3@b2 + b3. Exact because the reference has no
// nonlinearity between the three einsum layers.
// ---------------------------------------------------------------------------
__global__ __launch_bounds__(128)
void pn_fold_kernel(const float* __restrict__ W1, const float* __restrict__ b1,
                    const float* __restrict__ W2, const float* __restrict__ b2,
                    const float* __restrict__ W3, const float* __restrict__ b3,
                    _Float16* __restrict__ Weff, float* __restrict__ beff)
{
    __shared__ float sW2[64 * 64];
    __shared__ float sW1[64 * PN_CIN];
    const int t = threadIdx.x;
    for (int i = t; i < 64 * 64; i += 128)      sW2[i] = W2[i];
    for (int i = t; i < 64 * PN_CIN; i += 128)  sW1[i] = W1[i];
    __syncthreads();

    const int o = t;  // 0..127 output row
    float tr[64];     // (W3 @ W2) row o
#pragma unroll
    for (int m = 0; m < 64; ++m) {
        float a = 0.0f;
#pragma unroll
        for (int j = 0; j < 64; ++j) a += W3[o * 64 + j] * sW2[j * 64 + m];
        tr[m] = a;
    }
    for (int c = 0; c < PN_CIN; ++c) {
        float a = 0.0f;
#pragma unroll
        for (int m = 0; m < 64; ++m) a += tr[m] * sW1[m * PN_CIN + c];
        Weff[o * PN_KP + c] = (_Float16)a;
    }
    for (int c = PN_CIN; c < PN_KP; ++c) Weff[o * PN_KP + c] = (_Float16)0.0f;

    float bb = b3[o];
#pragma unroll
    for (int j = 0; j < 64; ++j) bb += W3[o * 64 + j] * b2[j];
#pragma unroll
    for (int m = 0; m < 64; ++m) bb += tr[m] * b1[m];
    beff[o] = bb;
}

// ---------------------------------------------------------------------------
// Kernel 3: ball query. One wave per (b,s). Ascending-index scan with
// ballot/popc prefix compaction reproduces the reference's sort-then-take-32
// semantics; pads with the first hit (centroid always hits itself).
// ---------------------------------------------------------------------------
__global__ __launch_bounds__(256)
void pn_ballq_kernel(const float* __restrict__ xyz,
                     const float* __restrict__ newxyz,
                     int* __restrict__ gidx)
{
    __shared__ int sIdx[8][PN_K];
    const int t = threadIdx.x;
    const int w = t >> 5;
    const int lane = t & 31;
    const int g = blockIdx.x * 8 + w;
    const int b = g >> 10;
    const int s = g & 1023;

    const float* cf = newxyz + (size_t)(b * PN_S + s) * 3;
    const float cx = cf[0], cy = cf[1], cz = cf[2];

    if (lane == 0) sIdx[w][0] = 0;

    int count = 0;
    for (int base = 0; base < PN_N && count < PN_K; base += 32) {
        const int p = base + lane;
        const float* px = xyz + ((size_t)b * PN_N + p) * 3;
        const float dx = px[0] - cx;
        const float dy = px[1] - cy;
        const float dz = px[2] - cz;
        const float d = dx * dx + dy * dy + dz * dz;
        const bool in = (d <= PN_R2);
        const unsigned mask = (unsigned)__ballot(in);
        const int pos = count + __popc(mask & ((1u << lane) - 1u));
        if (in && pos < PN_K) sIdx[w][pos] = p;
        count += __popc(mask);
    }
    const int cnt = count > PN_K ? PN_K : count;
    const int myidx = (lane < cnt) ? sIdx[w][lane] : sIdx[w][0];
    gidx[g * PN_K + lane] = myidx;
}

// ---------------------------------------------------------------------------
// Kernel 4: fused gather + GEMM (WMMA f16) + column-max + bias.
// 128 threads = 4 waves; each wave owns one sample s:
//   D(128x32) = W_eff(128x96) @ G(96x32), then out[row] = max_col D + b_eff.
// W_eff staged once per block to LDS (24 KB); G packed per wave to LDS in
// B-fragment layout (column-major K, 6 KB/wave). 48 v_wmma per wave.
// ---------------------------------------------------------------------------
__global__ __launch_bounds__(128)
void pn_mlp_max_kernel(const float* __restrict__ xyz,
                       const float* __restrict__ feats,
                       const float* __restrict__ newxyz,
                       const int* __restrict__ gidx,
                       const _Float16* __restrict__ Weff,
                       const float* __restrict__ beff,
                       float* __restrict__ outf)
{
    __shared__ _Float16 sW[PN_M * PN_KP];          // 24 KB
    __shared__ _Float16 sG[4][PN_K * PN_KP];       // 4 x 6 KB

    const int t = threadIdx.x;
    const int w = t >> 5;
    const int lane = t & 31;

    // Stage W_eff (f16) into LDS, 32-bit copies.
    {
        const unsigned* src = (const unsigned*)Weff;
        unsigned* dst = (unsigned*)sW;
        for (int i = t; i < (PN_M * PN_KP) / 2; i += 128) dst[i] = src[i];
    }

    const int g = blockIdx.x * 4 + w;
    const int b = g >> 10;
    const int s = g & 1023;

    // Gather one neighbor column per lane into B-layout (column-major K).
    {
        const int j = gidx[g * PN_K + lane];
        const float* cf = newxyz + (size_t)(b * PN_S + s) * 3;
        const float cx = cf[0], cy = cf[1], cz = cf[2];
        const float* px = xyz + ((size_t)b * PN_N + j) * 3;
        _Float16* gcol = &sG[w][lane * PN_KP];
        gcol[0] = (_Float16)(px[0] - cx);
        gcol[1] = (_Float16)(px[1] - cy);
        gcol[2] = (_Float16)(px[2] - cz);
        const float* pf = feats + ((size_t)b * PN_N + j) * PN_C;
        __builtin_prefetch(pf, 0, 0);
#pragma unroll
        for (int d = 0; d < PN_C; d += 4) {
            const float4 v = *(const float4*)(pf + d);
            gcol[3 + d + 0] = (_Float16)v.x;
            gcol[3 + d + 1] = (_Float16)v.y;
            gcol[3 + d + 2] = (_Float16)v.z;
            gcol[3 + d + 3] = (_Float16)v.w;
        }
#pragma unroll
        for (int k2 = PN_CIN; k2 < PN_KP; ++k2) gcol[k2] = (_Float16)0.0f;
    }
    __syncthreads();

    v8f acc[8][2];
#pragma unroll
    for (int mt = 0; mt < 8; ++mt) {
#pragma unroll
        for (int nt = 0; nt < 2; ++nt) {
            acc[mt][nt] = (v8f){0.f, 0.f, 0.f, 0.f, 0.f, 0.f, 0.f, 0.f};
        }
    }

    const int rsel = (lane < 16) ? 0 : 1;   // lane half
    const int arow = lane & 15;

#pragma unroll
    for (int kc = 0; kc < 3; ++kc) {
        // B fragments: lane holds column (lane&15)+16*nt, K = kc*32 + 16*rsel .. +15
        const int kbB = kc * 32 + rsel * 16;
        const v16h bf0 = *(const v16h*)&sG[w][(arow)      * PN_KP + kbB];
        const v16h bf1 = *(const v16h*)&sG[w][(arow + 16) * PN_KP + kbB];
        const int kbA = kc * 32 + rsel * 16;
#pragma unroll
        for (int mt = 0; mt < 8; ++mt) {
            // A fragment: row mt*16 + (lane&15), K = kbA..kbA+15
            const _Float16* ap = &sW[(mt * 16 + arow) * PN_KP + kbA];
            union { v16h v; v8h h[2]; } a;
            a.h[0] = *(const v8h*)(ap);
            a.h[1] = *(const v8h*)(ap + 8);
            acc[mt][0] = __builtin_amdgcn_wmma_f32_16x16x32_f16(
                false, a.v, false, bf0, (short)0, acc[mt][0], false, false);
            acc[mt][1] = __builtin_amdgcn_wmma_f32_16x16x32_f16(
                false, a.v, false, bf1, (short)0, acc[mt][1], false, false);
        }
    }

    // max over the 32 columns (two N-tiles merged, then 16-lane shfl tree),
    // bias added after the max (exact: bias is per-row constant).
#pragma unroll
    for (int mt = 0; mt < 8; ++mt) {
#pragma unroll
        for (int v = 0; v < 8; ++v) {
            float m = fmaxf(acc[mt][0][v], acc[mt][1][v]);
            m = fmaxf(m, __shfl_xor(m, 1));
            m = fmaxf(m, __shfl_xor(m, 2));
            m = fmaxf(m, __shfl_xor(m, 4));
            m = fmaxf(m, __shfl_xor(m, 8));
            const int row = mt * 16 + v + rsel * 8;
            if (arow == 0) {
                outf[((size_t)b * PN_M + row) * PN_S + s] = m + beff[row];
            }
        }
    }
}

// ---------------------------------------------------------------------------
// Host launcher
// ---------------------------------------------------------------------------
extern "C" void kernel_launch(void* const* d_in, const int* in_sizes, int n_in,
                              void* d_out, int out_size, void* d_ws, size_t ws_size,
                              hipStream_t stream) {
    const float* xyz      = (const float*)d_in[0];
    const float* features = (const float*)d_in[1];
    const float* W1 = (const float*)d_in[2];
    const float* b1 = (const float*)d_in[3];
    const float* W2 = (const float*)d_in[4];
    const float* b2 = (const float*)d_in[5];
    const float* W3 = (const float*)d_in[6];
    const float* b3 = (const float*)d_in[7];

    // Outputs concatenated flat in return order: new_xyz, new_features, fps_idx
    float* out_xyz  = (float*)d_out;                       // 16*1024*3
    float* out_feat = out_xyz + PN_B * PN_S * 3;           // 16*128*1024
    int*   out_idx  = (int*)(out_feat + (size_t)PN_B * PN_M * PN_S);

    // Workspace: group indices, folded weights (f16), folded bias (f32)
    int*      ws_idx = (int*)d_ws;                                   // 16384*32 ints
    _Float16* ws_W   = (_Float16*)((char*)d_ws + (size_t)PN_B * PN_S * PN_K * sizeof(int));
    float*    ws_b   = (float*)((char*)ws_W + (size_t)PN_M * PN_KP * sizeof(_Float16));

    pn_fps_kernel<<<PN_B, 1024, 0, stream>>>(xyz, out_xyz, out_idx);
    pn_fold_kernel<<<1, 128, 0, stream>>>(W1, b1, W2, b2, W3, b3, ws_W, ws_b);
    pn_ballq_kernel<<<(PN_B * PN_S) / 8, 256, 0, stream>>>(xyz, out_xyz, ws_idx);
    pn_mlp_max_kernel<<<(PN_B * PN_S) / 4, 128, 0, stream>>>(
        xyz, features, out_xyz, ws_idx, ws_W, ws_b, out_feat);
}
